// GPUOptimizedSubdomainClassifier_58875411693872
// MI455X (gfx1250) — compile-verified, matrix operand order checked by reading, not run on hardware
//
#include <hip/hip_runtime.h>
#include <hip/hip_bf16.h>

typedef __attribute__((ext_vector_type(16))) __bf16 v16bf;
typedef __attribute__((ext_vector_type(8)))  float  v8f;

union BFrag { unsigned u[8]; uint4 q[2]; v16bf v; };

// single f32 -> bf16 (RNE), cold staging path only
__device__ __forceinline__ unsigned short f2bf(float f) {
  unsigned u = __float_as_uint(f);
  u += 0x7FFFu + ((u >> 16) & 1u);
  return (unsigned short)(u >> 16);
}

// two f32 -> packed bf16x2 in ONE v_perm_b32 (+2 adds for round-half-up)
__device__ __forceinline__ unsigned pk2bf(float a, float b) {
  unsigned ua = __float_as_uint(a) + 0x8000u;
  unsigned ub = __float_as_uint(b) + 0x8000u;
  return __builtin_amdgcn_perm(ub, ua, 0x07060302u);  // [ua.hi16, ub.hi16]
}

// ---- LDS layout (bytes). Padded row strides (odd multiples of 16B) give
// ---- conflict-free banking without per-access XOR math. Total 324,208 < 320KB.
#define S_IM2   40    // bf16 row stride for [256 px][32 k]
#define S_W1    40    // bf16 row stride for [256 o][32 k]
#define S_256   264   // bf16 row stride for 256-deep rows (feat, w2)
#define S_128   136   // bf16 row stride for 128-deep rows (h, w3)

#define LDS_XS     0        // float[3][18][18]          3888 B
#define LDS_IM2    3888     // bf16 [256][S_IM2]        20480 B
#define LDS_W1B    24368    // bf16 [256][S_W1]         20480 B
#define LDS_SCALE  44848    // float[256]                1024 B
#define LDS_SHIFT  45872    // float[256]                1024 B
#define LDS_B2     46896    // float[128]                 512 B
#define LDS_B3     47408    // float[16]                   64 B
#define LDS_W2B    47472    // bf16 [128][S_256]        67584 B
#define LDS_W3B    115056   // bf16 [16][S_128]          4352 B
#define LDS_FEAT   119408   // bf16 [256][S_256]       135168 B (reused: logits+partials)
#define LDS_H      254576   // bf16 [256][S_128]        69632 B
#define LDS_TOTAL  324208

__device__ __forceinline__ void ldfrag(BFrag& f, const unsigned short* p) {
  f.q[0] = *(const uint4*)(p);
  f.q[1] = *(const uint4*)(p + 16);
}

extern "C" __global__ __launch_bounds__(256, 1)
void fused_tile_kernel(const float* __restrict__ x,  const float* __restrict__ w1,
                       const float* __restrict__ b1, const float* __restrict__ bng,
                       const float* __restrict__ bnb,const float* __restrict__ bnm,
                       const float* __restrict__ bnv,const float* __restrict__ w2,
                       const float* __restrict__ b2, const float* __restrict__ w3,
                       const float* __restrict__ b3, float* __restrict__ pooled)
{
  extern __shared__ char smem[];
  float*          xs  = (float*)(smem + LDS_XS);
  unsigned short* im2 = (unsigned short*)(smem + LDS_IM2);
  unsigned short* w1b = (unsigned short*)(smem + LDS_W1B);
  float*          scl = (float*)(smem + LDS_SCALE);
  float*          sh2 = (float*)(smem + LDS_SHIFT);
  float*          b2s = (float*)(smem + LDS_B2);
  float*          b3s = (float*)(smem + LDS_B3);
  unsigned short* w2b = (unsigned short*)(smem + LDS_W2B);
  unsigned short* w3b = (unsigned short*)(smem + LDS_W3B);
  unsigned short* ft  = (unsigned short*)(smem + LDS_FEAT);
  float*          lg  = (float*)(smem + LDS_FEAT);            // [256 px][16 o], reuse
  float*          part= (float*)(smem + LDS_FEAT + 16384);    // [256], reuse
  unsigned short* hb  = (unsigned short*)(smem + LDS_H);

  const int tid  = threadIdx.x;
  const int lane = tid & 31;
  const int wv   = __builtin_amdgcn_readfirstlane(tid >> 5);  // wave-uniform -> SGPR
  const int tx = blockIdx.x, ty = blockIdx.y, b = blockIdx.z;

  // ---------------- phase 1: stage inputs / weights into LDS ----------------
  for (int i = tid; i < 972; i += 256) {               // 18x18x3 halo, SAME pad
    int ci = i / 324, r = i - ci * 324;
    int hy = r / 18, hx = r - hy * 18;
    int gy = ty * 16 + hy - 1, gx = tx * 16 + hx - 1;
    float v = 0.f;
    if (gy >= 0 && gy < 512 && gx >= 0 && gx < 512)
      v = x[((b * 3 + ci) * 512 + gy) * 512 + gx];
    xs[i] = v;
  }
  for (int i = tid; i < 256 * 32; i += 256) {          // w1 -> bf16, K 27->32 pad
    int o = i >> 5, j = i & 31;
    w1b[o * S_W1 + j] = (j < 27) ? f2bf(w1[o * 27 + j]) : (unsigned short)0;
  }
  {                                                    // fold conv bias + BN
    float s = bng[tid] * rsqrtf(bnv[tid] + 1e-5f);
    scl[tid] = s;
    sh2[tid] = b1[tid] * s + bnb[tid] - bnm[tid] * s;
  }
  for (int i = tid; i < 128 * 256; i += 256) {
    int o = i >> 8, c = i & 255;
    w2b[o * S_256 + c] = f2bf(w2[i]);
  }
  for (int i = tid; i < 16 * 128; i += 256) {
    int o = i >> 7, c = i & 127;
    w3b[o * S_128 + c] = f2bf(w3[i]);
  }
  if (tid < 128) b2s[tid] = b2[tid];
  if (tid < 16)  b3s[tid] = b3[tid];
  __syncthreads();

  // ---------------- phase 2: im2col (one pixel per thread) ------------------
  {
    int py = tid >> 4, px = tid & 15;
    unsigned short* row = im2 + tid * S_IM2;
    #pragma unroll
    for (int ci = 0; ci < 3; ++ci)
      #pragma unroll
      for (int dy = 0; dy < 3; ++dy)
        #pragma unroll
        for (int dx = 0; dx < 3; ++dx)
          row[ci * 9 + dy * 3 + dx] =
              f2bf(xs[ci * 324 + (py + dy) * 18 + (px + dx)]);
    #pragma unroll
    for (int j = 27; j < 32; ++j) row[j] = 0;
  }
  __syncthreads();

  const int m = lane & 15, g = lane >> 4;
  const int kg = 8 * g;                 // half-wave K offset in fragments

  // ------------- GEMM0: conv as [256o x 32k] x [32k x 256px], BN+ReLU -------
  for (int mt = wv; mt < 16; mt += 8) {
    const int o0 = mt * 16 + m;
    const int ob = mt * 16 + kg;        // this lane's 8 consecutive out channels
    BFrag a;
    ldfrag(a, w1b + o0 * S_W1 + kg);
    float sc[8], sf[8];
    *(float4*)(sc)     = *(const float4*)(scl + ob);
    *(float4*)(sc + 4) = *(const float4*)(scl + ob + 4);
    *(float4*)(sf)     = *(const float4*)(sh2 + ob);
    *(float4*)(sf + 4) = *(const float4*)(sh2 + ob + 4);

    BFrag bf[2];
    ldfrag(bf[0], im2 + m * S_IM2 + kg);              // prefetch nt = 0
    #pragma unroll
    for (int nt = 0; nt < 16; ++nt) {
      const int cur = nt & 1;
      if (nt < 15)                                    // prefetch nt+1 before use
        ldfrag(bf[cur ^ 1], im2 + ((nt + 1) * 16 + m) * S_IM2 + kg);
      const int pxc = nt * 16 + m;
      v8f acc = {0.f,0.f,0.f,0.f,0.f,0.f,0.f,0.f};
      acc = __builtin_amdgcn_wmma_f32_16x16x32_bf16(false, a.v, false, bf[cur].v,
                                                    (short)0, acc, false, false);
      uint4 pk;
      pk.x = pk2bf(fmaxf(acc[0]*sc[0]+sf[0],0.f), fmaxf(acc[1]*sc[1]+sf[1],0.f));
      pk.y = pk2bf(fmaxf(acc[2]*sc[2]+sf[2],0.f), fmaxf(acc[3]*sc[3]+sf[3],0.f));
      pk.z = pk2bf(fmaxf(acc[4]*sc[4]+sf[4],0.f), fmaxf(acc[5]*sc[5]+sf[5],0.f));
      pk.w = pk2bf(fmaxf(acc[6]*sc[6]+sf[6],0.f), fmaxf(acc[7]*sc[7]+sf[7],0.f));
      *(uint4*)(ft + pxc * S_256 + ob) = pk;          // one ds_store_b128
    }
  }
  __syncthreads();

  // ------------- GEMM1: [128o x 256k] x [256k x 256px], bias+ReLU -----------
  {
    const int o0 = wv * 16 + m;
    const int ob = wv * 16 + kg;
    const unsigned short* arow = w2b + o0 * S_256 + kg;
    float bb[8];
    *(float4*)(bb)     = *(const float4*)(b2s + ob);
    *(float4*)(bb + 4) = *(const float4*)(b2s + ob + 4);
    for (int nb = 0; nb < 4; ++nb) {                  // 4 N-tiles share each A frag
      v8f acc[4];
      #pragma unroll
      for (int j = 0; j < 4; ++j) acc[j] = (v8f){0.f,0.f,0.f,0.f,0.f,0.f,0.f,0.f};

      BFrag a[2], bf[2][4];                           // double-buffered K pipeline
      ldfrag(a[0], arow);
      #pragma unroll
      for (int j = 0; j < 4; ++j)
        ldfrag(bf[0][j], ft + ((nb * 4 + j) * 16 + m) * S_256 + kg);

      #pragma unroll
      for (int kb = 0; kb < 8; ++kb) {
        const int cur = kb & 1, nxt = cur ^ 1;
        if (kb < 7) {                                 // prefetch kb+1
          ldfrag(a[nxt], arow + (kb + 1) * 32);
          #pragma unroll
          for (int j = 0; j < 4; ++j)
            ldfrag(bf[nxt][j],
                   ft + ((nb * 4 + j) * 16 + m) * S_256 + (kb + 1) * 32 + kg);
        }
        #pragma unroll
        for (int j = 0; j < 4; ++j)
          acc[j] = __builtin_amdgcn_wmma_f32_16x16x32_bf16(
              false, a[cur].v, false, bf[cur][j].v, (short)0, acc[j], false, false);
      }
      #pragma unroll
      for (int j = 0; j < 4; ++j) {
        const int pxc = (nb * 4 + j) * 16 + m;
        uint4 pk;
        pk.x = pk2bf(fmaxf(acc[j][0]+bb[0],0.f), fmaxf(acc[j][1]+bb[1],0.f));
        pk.y = pk2bf(fmaxf(acc[j][2]+bb[2],0.f), fmaxf(acc[j][3]+bb[3],0.f));
        pk.z = pk2bf(fmaxf(acc[j][4]+bb[4],0.f), fmaxf(acc[j][5]+bb[5],0.f));
        pk.w = pk2bf(fmaxf(acc[j][6]+bb[6],0.f), fmaxf(acc[j][7]+bb[7],0.f));
        *(uint4*)(hb + pxc * S_128 + ob) = pk;
      }
    }
  }
  __syncthreads();

  // ------------- GEMM2: [16o x 128k] x [128k x 256px] + bias ----------------
  {
    float bv[8];
    *(float4*)(bv)     = *(const float4*)(b3s + kg);
    *(float4*)(bv + 4) = *(const float4*)(b3s + kg + 4);
    const unsigned short* arow = w3b + m * S_128 + kg;
    for (int nt = wv * 2; nt < wv * 2 + 2; ++nt) {
      const int pxc = nt * 16 + m;
      const unsigned short* brow = hb + pxc * S_128 + kg;
      v8f acc = {0.f,0.f,0.f,0.f,0.f,0.f,0.f,0.f};
      BFrag a[2], bf[2];
      ldfrag(a[0], arow);
      ldfrag(bf[0], brow);
      #pragma unroll
      for (int kb = 0; kb < 4; ++kb) {
        const int cur = kb & 1, nxt = cur ^ 1;
        if (kb < 3) {
          ldfrag(a[nxt],  arow + (kb + 1) * 32);
          ldfrag(bf[nxt], brow + (kb + 1) * 32);
        }
        acc = __builtin_amdgcn_wmma_f32_16x16x32_bf16(false, a[cur].v, false, bf[cur].v,
                                                      (short)0, acc, false, false);
      }
      float4 lo = make_float4(acc[0]+bv[0], acc[1]+bv[1], acc[2]+bv[2], acc[3]+bv[3]);
      float4 hi = make_float4(acc[4]+bv[4], acc[5]+bv[5], acc[6]+bv[6], acc[7]+bv[7]);
      *(float4*)(lg + pxc * 16 + kg)     = lo;        // logits [px][16]
      *(float4*)(lg + pxc * 16 + kg + 4) = hi;
    }
  }
  __syncthreads();

  // ------------- 16x16 tile mean -> pooled[b, 16, ty*32+tx] -----------------
  {
    const int o = tid & 15, j = tid >> 4;
    float s = 0.f;
    #pragma unroll
    for (int i = 0; i < 16; ++i) s += lg[(j * 16 + i) * 16 + o];
    part[tid] = s;
  }
  __syncthreads();
  if (tid < 16) {
    float s = 0.f;
    #pragma unroll
    for (int j = 0; j < 16; ++j) s += part[j * 16 + tid];
    pooled[((b * 16 + tid) << 10) + ty * 32 + tx] = s * (1.f / 256.f);
  }
}

// out[b,c,h,w] = pooled[b,c,2h]  (64 MB broadcast, float4 stores)
extern "C" __global__ __launch_bounds__(128)
void bcast_kernel(const float* __restrict__ pooled, float* __restrict__ out)
{
  int row = blockIdx.x;              // = ((b*16)+c)*512 + h
  int h = row & 511;
  int c = (row >> 9) & 15;
  int b = row >> 13;
  float v = pooled[((b * 16 + c) << 10) + (h << 1)];
  float4 f4 = make_float4(v, v, v, v);
  ((float4*)(out + (size_t)row * 512))[threadIdx.x] = f4;
}

extern "C" void kernel_launch(void* const* d_in, const int* in_sizes, int n_in,
                              void* d_out, int out_size, void* d_ws, size_t ws_size,
                              hipStream_t stream)
{
  const float* x   = (const float*)d_in[0];
  const float* w1  = (const float*)d_in[1];
  const float* b1  = (const float*)d_in[2];
  const float* bng = (const float*)d_in[3];
  const float* bnb = (const float*)d_in[4];
  const float* bnm = (const float*)d_in[5];
  const float* bnv = (const float*)d_in[6];
  const float* w2  = (const float*)d_in[7];
  const float* b2  = (const float*)d_in[8];
  const float* w3  = (const float*)d_in[9];
  const float* b3  = (const float*)d_in[10];
  float* pooled = (float*)d_ws;                 // 4*16*1024 floats = 256 KB

  dim3 grid(32, 32, 4);                         // one 16x16 tile per block
  fused_tile_kernel<<<grid, 256, LDS_TOTAL, stream>>>(
      x, w1, b1, bng, bnb, bnm, bnv, w2, b2, w3, b3, pooled);

  bcast_kernel<<<dim3(4 * 16 * 512), 128, 0, stream>>>(pooled, (float*)d_out);
}
